// Deformable_Conv2d_68478958567441
// MI455X (gfx1250) — matrix operand order repaired and checked by reading, not compiled, not run on hardware
//
#include <hip/hip_runtime.h>
#include <hip/hip_bf16.h>

// ---------------- problem constants ----------------
#define B_    2
#define CIN   64
#define COUT  64
#define H_    128
#define W_    128
#define NTAP  9            // 3x3 taps
#define KTOT  576          // CIN * NTAP
#define NCHUNK 18          // KTOT / 32
#define HP_   130          // H + 2*PAD
// aux conv output channels: 18 offset + 9 mask + 3 adaptive = 30 (padded to 32)

typedef __attribute__((ext_vector_type(16))) __bf16 v16bf;
typedef __attribute__((ext_vector_type(8)))  float  v8f;
typedef __attribute__((ext_vector_type(4)))  unsigned int u32x4;
typedef __attribute__((ext_vector_type(8)))  unsigned int u32x8;

union ABu { unsigned int u[8]; v16bf v; };

__device__ __forceinline__ unsigned short f2bf(float f) {
    unsigned int u = __float_as_uint(f);
    unsigned int r = (u + 0x7FFFu + ((u >> 16) & 1u)) >> 16;   // RNE
    return (unsigned short)r;
}
__device__ __forceinline__ float sigmoidf_(float x) {
    return 1.0f / (1.0f + __expf(-x));
}
__device__ __forceinline__ int clampi(int v, int lo, int hi) {
    return v < lo ? lo : (v > hi ? hi : v);
}

// ---------------------------------------------------------------------------
// Kernel 0: repack weights to bf16 in WMMA-B friendly [n][k] layout,
//           k = tap*64 + ci  (pairs of consecutive k load as one b32).
// wb1: 32 x 576 (18 offset ch, 9 mask ch, 3 adaptive ch, 2 zero pad)
// wb2: 64 x 576 (final conv)
// ---------------------------------------------------------------------------
__global__ void pack_weights(const float* __restrict__ wp,
                             const float* __restrict__ wm,
                             const float* __restrict__ wad,
                             const float* __restrict__ wc,
                             unsigned short* __restrict__ wb1,
                             unsigned short* __restrict__ wb2) {
    int idx = blockIdx.x * blockDim.x + threadIdx.x;
    const int n1 = 32 * KTOT;
    if (idx < n1) {
        int n = idx / KTOT, k = idx % KTOT;
        int tap = k >> 6, ci = k & 63;
        float v = 0.f;
        if (n < 18)       v = wp [(n     ) * KTOT + ci * NTAP + tap];
        else if (n < 27)  v = wm [(n - 18) * KTOT + ci * NTAP + tap];
        else if (n < 30)  v = wad[(n - 27) * KTOT + ci * NTAP + tap];
        wb1[idx] = f2bf(v);
    } else {
        int j = idx - n1;
        if (j < COUT * KTOT) {
            int n = j / KTOT, k = j % KTOT;
            int tap = k >> 6, ci = k & 63;
            wb2[j] = f2bf(wc[n * KTOT + ci * NTAP + tap]);
        }
    }
}

// ---------------------------------------------------------------------------
// Kernel 1: fused aux convs (offset/mask/adaptive) via bf16 WMMA implicit GEMM.
// x patch staged global->LDS with ASYNC loads (ASYNCcnt), bulk-converted to
// bf16 once, then a clean ds_load_b128 -> v_wmma pipeline. Epilogue produces
// per-(pixel,tap) sampling coords p_x,p_y and modulation m*ad_m.
// Workgroup: 128 threads (4 waves), 32-pixel row segment. M-tiles=2, N-tiles=2.
// ---------------------------------------------------------------------------
__global__ __launch_bounds__(128)
void aux_conv(const float* __restrict__ x,
              const float* __restrict__ b_p,
              const unsigned short* __restrict__ wb1,
              float* __restrict__ px_buf,
              float* __restrict__ py_buf,
              float* __restrict__ mm_buf) {
    __shared__ float          patch_f [3 * 34 * CIN]; // [kh][col(34)][ci] f32
    __shared__ unsigned short patch_bf[3 * 34 * CIN]; // same layout, bf16
    __shared__ float          conv[32 * 32];          // [pixel][channel] f32

    const int seg = blockIdx.x & 3;
    const int h   = (blockIdx.x >> 2) & 127;
    const int b   = blockIdx.x >> 9;
    const int w0  = seg * 32;
    const int tid = threadIdx.x;

    // ---- stage 3x34x64 input patch via async global->LDS (zero padded) ----
    for (int i = tid; i < 3 * 34 * CIN; i += 128) {
        int ci = i & 63;
        int c  = (i >> 6) % 34;
        int kh = i / (34 * CIN);
        int gw = w0 - 1 + c;
        int gh = h - 1 + kh;
        unsigned ldsa = (unsigned)(size_t)&patch_f[i];
        if (gh >= 0 && gh < H_ && gw >= 0 && gw < W_) {
            unsigned goff = (unsigned)(((((b * CIN + ci) * H_) + gh) * W_ + gw) * 4);
            asm volatile("global_load_async_to_lds_b32 %0, %1, %2"
                         :: "v"(ldsa), "v"(goff), "s"(x) : "memory");
        } else {
            patch_f[i] = 0.f;
        }
    }
    asm volatile("s_wait_asynccnt 0x0" ::: "memory");
    __syncthreads();

    // ---- bulk f32 -> bf16 conversion pass (keeps GEMM loop pure) ----
    for (int i = tid; i < 3 * 34 * CIN; i += 128)
        patch_bf[i] = f2bf(patch_f[i]);
    __syncthreads();

    const int wave = tid >> 5, lane = tid & 31;
    const int mtile = wave >> 1, ntile = wave & 1;
    const int hh = lane >> 4, ln = lane & 15;
    const unsigned int* wb1u = (const unsigned int*)wb1;

    v8f acc = {};
#pragma unroll
    for (int kc = 0; kc < NCHUNK; ++kc) {
        const int tap = kc >> 1;
        const int kh = tap / 3, kw = tap % 3;
        const int cib = (kc & 1) * 32;
        ABu a, bm;
#pragma unroll
        for (int v = 0; v < 8; ++v) {
            // A 16x32 bf16 layout: lanes 0-15 / 16-31, VGPR v -> K pairs
            int klo = ((v < 4) ? 2 * v : 16 + 2 * (v - 4)) + 8 * hh;
            int ci  = cib + klo;
            int col = mtile * 16 + ln + kw;
            a.u[v] = *(const unsigned int*)&patch_bf[(kh * 34 + col) * CIN + ci];
            // B 32x16 bf16 layout: same K order as A (VGPR v -> K pairs klo)
            int kb = kc * 32 + klo;
            int nb = ntile * 16 + ln;
            bm.u[v] = wb1u[(nb * KTOT + kb) >> 1];
        }
        acc = __builtin_amdgcn_wmma_f32_16x16x32_bf16(false, a.v, false, bm.v,
                                                      (short)0, acc, false, false);
    }
#pragma unroll
    for (int r = 0; r < 8; ++r)
        conv[(mtile * 16 + r + 8 * hh) * 32 + ntile * 16 + ln] = acc[r];
    __syncthreads();

    // ---- epilogue: sampling coordinates + modulation ----
    for (int item = tid; item < 32 * NTAP; item += 128) {
        int pix = item / NTAP, tap = item % NTAP;
        float offx = conv[pix * 32 + tap]      + b_p[tap];
        float offy = conv[pix * 32 + 9 + tap]  + b_p[9 + tap];
        float adb  = 1.f - sigmoidf_(conv[pix * 32 + 27 + (tap % 3)]);
        float ad   = 2.f * adb;                    // DIL * ad_base
        float adm  = 2.f * (adb - 0.5f) * 2.f;     // DIL * (ad_base-0.5)*2
        float mval = sigmoidf_(conv[pix * 32 + 18 + tap]);
        float pnx  = (float)(tap / 3) - 1.f;
        float pny  = (float)(tap % 3) - 1.f;
        int gw = w0 + pix;
        float px = (float)(h + 1)  + pnx + offx + ad * pnx;
        float py = (float)(gw + 1) + pny + offy + ad * pny;
        long idx = (((long)b * H_ * W_ + (long)h * W_ + gw) * NTAP + tap);
        px_buf[idx] = px;
        py_buf[idx] = py;
        mm_buf[idx] = mval * adm;
    }
}

// ---------------------------------------------------------------------------
// Kernel 2: TDM loads packed w_conv (72KB bf16) into LDS while the waves do
// bilinear gathers into the bf16 A tile; then 4 waves x 18 bf16 WMMAs.
// Workgroup: 128 threads (4 waves), 16-pixel row segment x 64 out channels.
// ---------------------------------------------------------------------------
__global__ __launch_bounds__(128)
void deform_gemm(const float* __restrict__ x,
                 const unsigned short* __restrict__ wb2,
                 const float* __restrict__ px_buf,
                 const float* __restrict__ py_buf,
                 const float* __restrict__ mm_buf,
                 float* __restrict__ out) {
    __shared__ unsigned short A[16 * KTOT];     // [pixel][k = tap*64+ci] bf16
    __shared__ unsigned short Wl[COUT * KTOT];  // packed w_conv tile (bf16)

    const int seg = blockIdx.x & 7;
    const int h   = (blockIdx.x >> 3) & 127;
    const int b   = blockIdx.x >> 10;
    const int w0  = seg * 16;
    const int tid = threadIdx.x;

    // ---- wave 0 kicks off the TDM weight DMA (overlaps sampling phase) ----
    const bool wave0 = (tid < 32);
    if (wave0) {
        const unsigned ne = (COUT * KTOT * 2) / 8;     // 9216 x 8-byte elements
        unsigned long long ga = (unsigned long long)(size_t)wb2;
        u32x4 g0;
        g0[0] = 1u;                                    // count=1, flags=0
        g0[1] = (unsigned)(size_t)&Wl[0];              // lds_addr
        g0[2] = (unsigned)(ga & 0xFFFFFFFFu);          // global_addr[31:0]
        g0[3] = (unsigned)((ga >> 32) & 0x01FFFFFFu)   // global_addr[56:32]
              | 0x80000000u;                           // type=2 (image)
        u32x8 g1;
        g1[0] = 3u << 16;                              // data_size=8B, mask=0
        g1[1] = (ne & 0xFFFFu) << 16;                  // tensor_dim0[15:0]
        g1[2] = (ne >> 16) | (1u << 16);               // tensor_dim0[31:16], tensor_dim1=1
        g1[3] = (ne & 0xFFFFu) << 16;                  // tile_dim0 = ne
        g1[4] = 0;                                     // tile_dim1/2 unused
        g1[5] = ne;                                    // tensor_dim0_stride[31:0]
        g1[6] = 0;
        g1[7] = 0;
        asm volatile("tensor_load_to_lds %0, %1" :: "s"(g0), "s"(g1) : "memory");
    }

    // ---- sampling phase: thread = (pixel, 8-channel group) ----
    const int pix = tid >> 3, sub = tid & 7;
    const int gw  = w0 + pix;
    const long pbase = (((long)b * H_ * W_ + (long)h * W_ + gw) * NTAP);
    const float* xb = x + (long)b * CIN * H_ * W_;
    __builtin_prefetch(px_buf + pbase, 0, 0);          // global_prefetch

    for (int tap = 0; tap < NTAP; ++tap) {
        float px = px_buf[pbase + tap];
        float py = py_buf[pbase + tap];
        float mm = mm_buf[pbase + tap];
        float fx = floorf(px), fy = floorf(py);
        int qltx = clampi((int)fx,     0, HP_ - 1);
        int qrbx = clampi((int)fx + 1, 0, HP_ - 1);
        int qlty = clampi((int)fy,     0, HP_ - 1);
        int qrby = clampi((int)fy + 1, 0, HP_ - 1);
        // mask / snap-to-floor / clamp (reference semantics)
        bool mx = (px < 1.f) || (px > 128.f);
        bool my = (py < 1.f) || (py > 128.f);
        float pxs = mx ? fx : px;  pxs = fminf(fmaxf(pxs, 0.f), (float)(HP_ - 1));
        float pys = my ? fy : py;  pys = fminf(fmaxf(pys, 0.f), (float)(HP_ - 1));
        float wltx = 1.f + ((float)qltx - pxs);
        float wrbx = 1.f - ((float)qrbx - pxs);
        float wlty = 1.f + ((float)qlty - pys);
        float wrby = 1.f - ((float)qrby - pys);
        float glt = wltx * wlty, grb = wrbx * wrby;
        float glb = wltx * wrby, grt = wrbx * wlty;
        // padded coords -> raw coords; pad region reads zero
        int rlt = qltx - 1, rrb = qrbx - 1, clt = qlty - 1, crb = qrby - 1;
        bool vr0 = (rlt >= 0) && (rlt < H_);
        bool vr1 = (rrb >= 0) && (rrb < H_);
        bool vc0 = (clt >= 0) && (clt < W_);
        bool vc1 = (crb >= 0) && (crb < W_);
#pragma unroll
        for (int cg = 0; cg < 8; ++cg) {
            int ci = sub * 8 + cg;
            const float* xc = xb + (long)ci * H_ * W_;
            float vlt = (vr0 && vc0) ? xc[rlt * W_ + clt] : 0.f;
            float vrb = (vr1 && vc1) ? xc[rrb * W_ + crb] : 0.f;
            float vlb = (vr0 && vc1) ? xc[rlt * W_ + crb] : 0.f;
            float vrt = (vr1 && vc0) ? xc[rrb * W_ + clt] : 0.f;
            float val = (glt * vlt + grb * vrb + glb * vlb + grt * vrt) * mm;
            A[pix * KTOT + tap * 64 + ci] = f2bf(val);
        }
    }
    if (wave0) {
        __builtin_amdgcn_s_wait_tensorcnt(0);          // weight DMA done
    }
    __syncthreads();

    // ---- GEMM phase: wave = N-tile (16 out channels), 18 K-chunks ----
    const int ntile = tid >> 5, lane = tid & 31;
    const int hh = lane >> 4, ln = lane & 15;
    const unsigned int* Au  = (const unsigned int*)A;
    const unsigned int* Wlu = (const unsigned int*)Wl;

    v8f acc = {};
#pragma unroll
    for (int kc = 0; kc < NCHUNK; ++kc) {
        ABu a, bm;
#pragma unroll
        for (int v = 0; v < 8; ++v) {
            int klo = ((v < 4) ? 2 * v : 16 + 2 * (v - 4)) + 8 * hh;
            a.u[v] = Au[(ln * KTOT + kc * 32 + klo) >> 1];
            int kb = kc * 32 + klo;
            bm.u[v] = Wlu[((ntile * 16 + ln) * KTOT + kb) >> 1];
        }
        acc = __builtin_amdgcn_wmma_f32_16x16x32_bf16(false, a.v, false, bm.v,
                                                      (short)0, acc, false, false);
    }

    const int co = ntile * 16 + ln;
#pragma unroll
    for (int r = 0; r < 8; ++r) {
        int pm = r + 8 * hh;     // pixel (M row) per C/D layout
        out[(((long)b * COUT + co) * H_ + h) * W_ + (w0 + pm)] = acc[r];
    }
}

// ---------------------------------------------------------------------------
extern "C" void kernel_launch(void* const* d_in, const int* in_sizes, int n_in,
                              void* d_out, int out_size, void* d_ws, size_t ws_size,
                              hipStream_t stream) {
    const float* x    = (const float*)d_in[0];
    const float* w_p  = (const float*)d_in[1];
    const float* b_p  = (const float*)d_in[2];
    const float* w_m  = (const float*)d_in[3];
    const float* w_ad = (const float*)d_in[4];
    const float* w_c  = (const float*)d_in[5];

    const size_t npt = (size_t)B_ * H_ * W_ * NTAP;   // per-(b,pixel,tap)
    float* px_buf = (float*)d_ws;
    float* py_buf = px_buf + npt;
    float* mm_buf = py_buf + npt;
    unsigned short* wb1 = (unsigned short*)(mm_buf + npt);
    unsigned short* wb2 = wb1 + 32 * KTOT;

    int pack_elems = 32 * KTOT + COUT * KTOT;
    pack_weights<<<(pack_elems + 255) / 256, 256, 0, stream>>>(
        w_p, w_m, w_ad, w_c, wb1, wb2);

    aux_conv<<<B_ * H_ * (W_ / 32), 128, 0, stream>>>(
        x, b_p, wb1, px_buf, py_buf, mm_buf);

    deform_gemm<<<B_ * H_ * (W_ / 16), 128, 0, stream>>>(
        x, wb2, px_buf, py_buf, mm_buf, (float*)d_out);
}
